// StageA_Simple_90056874262572
// MI455X (gfx1250) — matrix operand order repaired and checked by reading, not compiled, not run on hardware
//
#include <hip/hip_runtime.h>
#include <hip/hip_bf16.h>
#include <math.h>

// ---------------------------------------------------------------------------
// GNN edge-gated message passing on MI455X (gfx1250, wave32, WMMA bf16).
// All GEMMs use v_wmma_f32_16x16x32_bf16 (one 16x16 D tile per wave in the encoders, 8 per wave in the gate kernel).
// Gate kernel: A gathered as bf16 (pre-materialized h copy), B staged in LDS.
// ---------------------------------------------------------------------------

typedef __attribute__((ext_vector_type(16))) __bf16 v16bf;
typedef __attribute__((ext_vector_type(8)))  __bf16 v8bf;
typedef __attribute__((ext_vector_type(8)))  float  v8f;

static constexpr int NN = 50000;    // nodes
static constexpr int CC = 1024;     // input genes
static constexpr int HH = 128;      // hidden dim
static constexpr int EE = 1600000;  // edges
static constexpr int LAYERS = 2;

__device__ __forceinline__ v8f wmma_bf16(v16bf a, v16bf b, v8f c) {
    return __builtin_amdgcn_wmma_f32_16x16x32_bf16(
        /*neg_a=*/false, a, /*neg_b=*/false, b,
        /*c_mod=*/(short)0, c, /*reuse_a=*/false, /*reuse_b=*/false);
}

// A-operand (16x32 bf16) from row-major f32, converting on the fly.
// lanes 0-15 row M=lane&15 (kbase 0), lanes 16-31 same rows (kbase 8);
// VGPRs 0-3 = K kbase+{0..7}, VGPRs 4-7 = K kbase+{16..23}.
__device__ __forceinline__ v16bf load_a_f32(const float* __restrict__ rowp, int k0, int lane) {
    const int kb = (lane & 16) ? 8 : 0;
    v16bf a;
#pragma unroll
    for (int j = 0; j < 8; ++j) {
        const int k = k0 + ((j & 4) ? 16 : 0) + kb + 2 * (j & 3);
        const float2 f = *reinterpret_cast<const float2*>(rowp + k);
        a[2 * j]     = (__bf16)f.x;
        a[2 * j + 1] = (__bf16)f.y;
    }
    return a;
}

// A-operand from row-major bf16: two aligned 16B loads map 1:1 onto lane layout.
__device__ __forceinline__ v16bf load_a_bf16(const __bf16* __restrict__ rowp, int k0, int lane) {
    const int kb = (lane & 16) ? 8 : 0;
    const v8bf lo = *reinterpret_cast<const v8bf*>(rowp + k0 + kb);       // K kb..kb+7
    const v8bf hi = *reinterpret_cast<const v8bf*>(rowp + k0 + 16 + kb);  // K 16+kb..
    v16bf a;
#pragma unroll
    for (int i = 0; i < 8; ++i) { a[i] = lo[i]; a[i + 8] = hi[i]; }
    return a;
}

// ---------------------------------------------------------------------------
// Pre-pack a row-major f32 weight [K,128] into per-lane bf16 B-operand blocks:
// block (kc,nt) = 32 lanes x 16 bf16 contiguous; lanes 0-15: K=kc*32+{0..7,16..23},
// lanes 16-31: K=kc*32+{8..15,24..31}; col = nt*16+(lane&15).
// ---------------------------------------------------------------------------
__global__ void pack_b_kernel(const float* __restrict__ W, __bf16* __restrict__ Wp, int K) {
    const int t = blockIdx.x * blockDim.x + threadIdx.x;
    const int total = (K >> 5) * 8 * 32;
    if (t >= total) return;
    const int lane = t & 31;
    const int nt   = (t >> 5) & 7;
    const int kc   = t >> 8;
    const int n    = nt * 16 + (lane & 15);
    const int koff = (lane & 16) ? 8 : 0;
    __bf16* out = Wp + (size_t)t * 16;
#pragma unroll
    for (int j = 0; j < 8; ++j) {
        const int k = kc * 32 + koff + ((j & 4) ? 16 : 0) + 2 * (j & 3);
        out[2 * j]     = (__bf16)W[(size_t)k * HH + n];
        out[2 * j + 1] = (__bf16)W[(size_t)(k + 1) * HH + n];
    }
}

__global__ void rho_kernel(const float* __restrict__ raw, float* __restrict__ rho, int n) {
    const int i = blockIdx.x * blockDim.x + threadIdx.x;
    if (i < n) rho[i] = 1.0f / (1.0f + __expf(-raw[i]));
}

__global__ void zero_kernel(float* __restrict__ p, long n) {
    const long i = (long)blockIdx.x * blockDim.x + threadIdx.x;
    if (i < n) p[i] = 0.0f;
}

// ---------------------------------------------------------------------------
// Encoder layer 1: H1 = relu(X @ W1 + b1)   [50000,1024]x[1024,128] -> bf16
// ---------------------------------------------------------------------------
__global__ void __launch_bounds__(128)
enc1_kernel(const float* __restrict__ X, const __bf16* __restrict__ W1p,
            const float* __restrict__ b1, __bf16* __restrict__ H1) {
    const int lane = threadIdx.x & 31;
    const int wid  = blockIdx.x * 4 + (threadIdx.x >> 5);
    const int mt = wid >> 3;      // 0..3124
    const int nt = wid & 7;       // 0..7
    const float* rowp = X + (size_t)(mt * 16 + (lane & 15)) * CC;
    v8f acc = {};
#pragma unroll 4
    for (int kc = 0; kc < CC / 32; ++kc) {
        const v16bf a = load_a_f32(rowp, kc * 32, lane);
        const v16bf b = *reinterpret_cast<const v16bf*>(
            W1p + (((size_t)kc * 8 + nt) * 32 + lane) * 16);
        acc = wmma_bf16(a, b, acc);
    }
    const int col = nt * 16 + (lane & 15);
    const float bias = b1[col];
    const int rbase = (lane & 16) ? 8 : 0;
#pragma unroll
    for (int r = 0; r < 8; ++r) {
        float v = acc[r] + bias;
        v = v > 0.0f ? v : 0.0f;
        H1[(size_t)(mt * 16 + rbase + r) * HH + col] = (__bf16)v;
    }
}

// ---------------------------------------------------------------------------
// Encoder layer 2: h = relu(H1 @ W2 + b2) -> f32 h AND bf16 hb (gate operand)
// ---------------------------------------------------------------------------
__global__ void __launch_bounds__(128)
enc2_kernel(const __bf16* __restrict__ H1, const __bf16* __restrict__ W2p,
            const float* __restrict__ b2, float* __restrict__ h, __bf16* __restrict__ hb) {
    const int lane = threadIdx.x & 31;
    const int wid  = blockIdx.x * 4 + (threadIdx.x >> 5);
    const int mt = wid >> 3;
    const int nt = wid & 7;
    const __bf16* rowp = H1 + (size_t)(mt * 16 + (lane & 15)) * HH;
    v8f acc = {};
#pragma unroll
    for (int kc = 0; kc < HH / 32; ++kc) {
        const v16bf a = load_a_bf16(rowp, kc * 32, lane);
        const v16bf b = *reinterpret_cast<const v16bf*>(
            W2p + (((size_t)kc * 8 + nt) * 32 + lane) * 16);
        acc = wmma_bf16(a, b, acc);
    }
    const int col = nt * 16 + (lane & 15);
    const float bias = b2[col];
    const int rbase = (lane & 16) ? 8 : 0;
#pragma unroll
    for (int r = 0; r < 8; ++r) {
        float v = acc[r] + bias;
        v = v > 0.0f ? v : 0.0f;
        const size_t idx = (size_t)(mt * 16 + rbase + r) * HH + col;
        h[idx]  = v;
        hb[idx] = (__bf16)v;
    }
}

// ---------------------------------------------------------------------------
// Edge gate + weighted scatter. One wave per 16 edges:
//   z = [hb[src], hb[dst]] (16 x 256 bf16)  -> WMMA vs Wg1 staged in LDS
//   g = sigmoid(relu(z @ Wg1 + bg1) @ Wg2 + bg2)
//   w = base_w * g * rho[src] * rho[dst]
//   m[dst,:] += w * h[src,:] (f32 atomics, L2-resident);  deg[dst] += w
// ---------------------------------------------------------------------------
__global__ void __launch_bounds__(128)
gate_scatter_kernel(const float* __restrict__ h, const __bf16* __restrict__ hb,
                    const int* __restrict__ src, const int* __restrict__ dst,
                    const float* __restrict__ base_w, const float* __restrict__ rho,
                    const __bf16* __restrict__ Wg1p, const float* __restrict__ bg1,
                    const float* __restrict__ Wg2, const float* __restrict__ bg2,
                    float* __restrict__ m, float* __restrict__ deg) {
    // Wg1 packed: 8 kc x 8 nt x 32 lanes x 16 bf16 = 64 KB, staged in LDS once.
    __shared__ __align__(16) __bf16 sB[8 * 8 * 32 * 16];
    __shared__ float s_g[4 * 16];
    __shared__ float s_w[4 * 16];
    __shared__ int   s_d[4 * 16];
    __shared__ int   s_s[4 * 16];

    // cooperative 64KB copy (4096 x uint4 across 128 threads)
    {
        const uint4* gsrc = reinterpret_cast<const uint4*>(Wg1p);
        uint4* gdst = reinterpret_cast<uint4*>(sB);
#pragma unroll
        for (int i = 0; i < 32; ++i)
            gdst[threadIdx.x + 128 * i] = gsrc[threadIdx.x + 128 * i];
    }
    __syncthreads();

    const int w    = threadIdx.x >> 5;
    const int lane = threadIdx.x & 31;
    const int tile = blockIdx.x * 4 + w;
    const int e0   = tile * 16;
    const int e    = e0 + (lane & 15);

    const int sn = src[e];
    const int dn = dst[e];
    const __bf16* hbs = hb + (size_t)sn * HH;
    const __bf16* hbd = hb + (size_t)dn * HH;

    v8f acc[8];
#pragma unroll
    for (int nt = 0; nt < 8; ++nt) acc[nt] = {};

#pragma unroll
    for (int kc = 0; kc < 8; ++kc) {           // K = 2H = 256
        const __bf16* rowp = (kc < 4) ? hbs : hbd;
        const v16bf a = load_a_bf16(rowp, (kc & 3) * 32, lane);
#pragma unroll
        for (int nt = 0; nt < 8; ++nt) {
            const v16bf b = *reinterpret_cast<const v16bf*>(
                sB + (((kc * 8 + nt) * 32 + lane) * 16));
            acc[nt] = wmma_bf16(a, b, acc[nt]);
        }
    }

    // epilogue: bias + relu + dot with Wg2; per-row partials per lane
    float part[8];
#pragma unroll
    for (int r = 0; r < 8; ++r) part[r] = 0.0f;
    const int cl = lane & 15;
#pragma unroll
    for (int nt = 0; nt < 8; ++nt) {
        const int n = nt * 16 + cl;
        const float bb = bg1[n];
        const float wv = Wg2[n];
#pragma unroll
        for (int r = 0; r < 8; ++r) {
            float v = acc[nt][r] + bb;
            v = v > 0.0f ? v : 0.0f;
            part[r] += v * wv;
        }
    }
    // reduce across the 16 lanes of each half (rows 0-7 / 8-15)
#pragma unroll
    for (int mask = 1; mask <= 8; mask <<= 1) {
#pragma unroll
        for (int r = 0; r < 8; ++r)
            part[r] += __shfl_xor(part[r], mask, 32);
    }
    if (lane == 0) {
#pragma unroll
        for (int r = 0; r < 8; ++r) s_g[w * 16 + r] = part[r];
    }
    if (lane == 16) {
#pragma unroll
        for (int r = 0; r < 8; ++r) s_g[w * 16 + 8 + r] = part[r];
    }
    __syncthreads();

    if (lane < 16) {
        const float gp   = s_g[w * 16 + lane] + bg2[0];
        const float gate = 1.0f / (1.0f + __expf(-gp));
        const float we   = base_w[e] * gate * rho[sn] * rho[dn];
        s_w[w * 16 + lane] = we;
        s_d[w * 16 + lane] = dn;
        s_s[w * 16 + lane] = sn;
        atomicAdd(&deg[dn], we);
    }
    __syncthreads();

    // scatter: lane owns 4 columns of the 128
    const int c0 = lane * 4;
    for (int i = 0; i < 16; ++i) {
        const float we  = s_w[w * 16 + i];
        const int   dn2 = s_d[w * 16 + i];
        const int   sn2 = s_s[w * 16 + i];
        const float4 hv = *reinterpret_cast<const float4*>(h + (size_t)sn2 * HH + c0);
        float* mp = m + (size_t)dn2 * HH + c0;
        atomicAdd(mp + 0, we * hv.x);
        atomicAdd(mp + 1, we * hv.y);
        atomicAdd(mp + 2, we * hv.z);
        atomicAdd(mp + 3, we * hv.w);
    }
}

// h = m / (deg + eps), plus bf16 mirror for the next layer's gate GEMM.
__global__ void normalize_kernel(const float* __restrict__ m, const float* __restrict__ deg,
                                 float* __restrict__ out, __bf16* __restrict__ outb) {
    const int i = blockIdx.x * blockDim.x + threadIdx.x;   // N*H threads exactly
    const int node = i >> 7;
    const float v = m[i] / (deg[node] + 1e-8f);
    out[i]  = v;
    outb[i] = (__bf16)v;
}

// ---------------------------------------------------------------------------
extern "C" void kernel_launch(void* const* d_in, const int* in_sizes, int n_in,
                              void* d_out, int out_size, void* d_ws, size_t ws_size,
                              hipStream_t stream) {
    const float* x       = (const float*)d_in[0];
    const int*   src     = (const int*)  d_in[1];
    const int*   dst     = (const int*)  d_in[2];
    const float* base_w  = (const float*)d_in[3];
    const float* W1      = (const float*)d_in[4];
    const float* b1      = (const float*)d_in[5];
    const float* W2      = (const float*)d_in[6];
    const float* b2      = (const float*)d_in[7];
    const float* Wg1     = (const float*)d_in[8];
    const float* bg1     = (const float*)d_in[9];
    const float* Wg2     = (const float*)d_in[10];
    const float* bg2     = (const float*)d_in[11];
    const float* rho_raw = (const float*)d_in[12];

    char* ws = (char*)d_ws;
    size_t off = 0;
    auto wsalloc = [&](size_t bytes) -> void* {
        void* p = ws + off;
        off += (bytes + 255) & ~(size_t)255;
        return p;
    };

    __bf16* W1p  = (__bf16*)wsalloc((size_t)CC * HH * 2);        // 256 KB
    __bf16* W2p  = (__bf16*)wsalloc((size_t)HH * HH * 2);        // 32 KB
    __bf16* Wg1p = (__bf16*)wsalloc((size_t)(2 * HH) * HH * 2);  // 64 KB
    float*  rho  = (float*) wsalloc((size_t)NN * 4);
    __bf16* H1   = (__bf16*)wsalloc((size_t)NN * HH * 2);        // 12.8 MB
    float*  h0   = (float*) wsalloc((size_t)NN * HH * 4);        // 25.6 MB
    float*  h1   = (float*) wsalloc((size_t)NN * HH * 4);        // 25.6 MB
    __bf16* hb0  = (__bf16*)wsalloc((size_t)NN * HH * 2);        // 12.8 MB
    __bf16* hb1  = (__bf16*)wsalloc((size_t)NN * HH * 2);        // 12.8 MB
    float*  mbuf = (float*) wsalloc((size_t)NN * HH * 4);        // 25.6 MB
    float*  degb = (float*) wsalloc((size_t)NN * 4);

    // Pre-pack weights into WMMA B-operand layout (bf16).
    pack_b_kernel<<<((CC / 32) * 8 * 32 + 255) / 256, 256, 0, stream>>>(W1, W1p, CC);
    pack_b_kernel<<<((HH / 32) * 8 * 32 + 255) / 256, 256, 0, stream>>>(W2, W2p, HH);
    pack_b_kernel<<<((2 * HH / 32) * 8 * 32 + 255) / 256, 256, 0, stream>>>(Wg1, Wg1p, 2 * HH);
    rho_kernel<<<(NN + 255) / 256, 256, 0, stream>>>(rho_raw, rho, NN);

    // Encoder: 3125 * 8 = 25000 tiles, 4 waves/block -> 6250 blocks.
    enc1_kernel<<<(NN / 16) * 8 / 4, 128, 0, stream>>>(x, W1p, b1, H1);
    enc2_kernel<<<(NN / 16) * 8 / 4, 128, 0, stream>>>(H1, W2p, b2, h0, hb0);

    const float*  hcur  = h0;
    const __bf16* hbcur = hb0;
    for (int l = 0; l < LAYERS; ++l) {
        zero_kernel<<<((long)NN * HH + 255) / 256, 256, 0, stream>>>(mbuf, (long)NN * HH);
        zero_kernel<<<(NN + 255) / 256, 256, 0, stream>>>(degb, (long)NN);
        // 100000 edge tiles, 4 waves/block -> 25000 blocks.
        gate_scatter_kernel<<<EE / 16 / 4, 128, 0, stream>>>(
            hcur, hbcur, src, dst, base_w, rho, Wg1p, bg1, Wg2, bg2, mbuf, degb);
        float*  hnext  = (l == LAYERS - 1) ? (float*)d_out : h1;
        __bf16* hbnext = (l == LAYERS - 1) ? hb0 /*unused*/ : hb1;
        normalize_kernel<<<(NN * HH) / 256, 256, 0, stream>>>(mbuf, degb, hnext, hbnext);
        hcur  = hnext;
        hbcur = hbnext;
    }
}
